// MLA_64261300683306
// MI455X (gfx1250) — matrix operand order repaired and checked by reading, not compiled, not run on hardware
//
#include <hip/hip_runtime.h>
#include <math.h>

typedef __bf16 bf16;
typedef bf16  bf16x8 __attribute__((ext_vector_type(8)));
typedef bf16  v16bf  __attribute__((ext_vector_type(16)));
typedef float v8f    __attribute__((ext_vector_type(8)));

#define H_DIM   2048
#define NHEADS  16
#define QLORA   1536
#define KVLORA  512
#define DNOPE   128
#define DROPE   64
#define DQK     192
#define DVAL    128
#define SEQ     2048
#define BATCH   2
#define MROWS   (BATCH * SEQ)
#define KV_OUT  (NHEADS * (DNOPE + DVAL))   // 4096
#define KV_IN   (KVLORA + DROPE)            // 576

// LDS byte offset of a __shared__ object: generic (flat) pointers to LDS carry
// the LDS offset in their low 32 bits (aperture in the high bits).
#define LDS_OFF(p) ((unsigned)(unsigned long long)(uintptr_t)(p))

// ---------------------------------------------------------------------------
// fp32 -> bf16 elementwise cast (grid-stride)
// ---------------------------------------------------------------------------
__global__ void cast_f32_to_bf16(const float* __restrict__ in,
                                 bf16* __restrict__ out, int n) {
    int i = blockIdx.x * blockDim.x + threadIdx.x;
    int stride = gridDim.x * blockDim.x;
    for (; i < n; i += stride) out[i] = (bf16)in[i];
}

// ---------------------------------------------------------------------------
// fragment pack: 8 + 8 bf16 -> v16bf (matches 16-bit A/B WMMA lane layout)
// ---------------------------------------------------------------------------
__device__ __forceinline__ v16bf pack16(const bf16* lo, const bf16* hi) {
    bf16x8 a = *(const bf16x8*)lo;
    bf16x8 b = *(const bf16x8*)hi;
    v16bf r;
#pragma unroll
    for (int e = 0; e < 8; e++) { r[e] = a[e]; r[8 + e] = b[e]; }
    return r;
}

// ---------------------------------------------------------------------------
// Tiled bf16 WMMA GEMM: C[M][N] = A[M][K] * B[N][K]^T
// 128x128 tile, K-step 32, 256 threads = 8 waves, wave = 32x64 subtile.
// Double-buffered LDS tiles filled by GLOBAL_LOAD_ASYNC_TO_LDS_B128 (ASYNCcnt)
// so the next tile's DMA overlaps the current tile's WMMAs.
// M % 128 == 0 and K % 32 == 0 required (true for all our calls); N ragged ok
// (out-of-range B rows clamp to row N-1; those columns are never stored).
// ---------------------------------------------------------------------------
template <typename OutT>
__global__ __launch_bounds__(256)
void gemm_bf16_wmma(const bf16* __restrict__ A, const bf16* __restrict__ B,
                    OutT* __restrict__ C, int M, int N, int K) {
    constexpr int BM = 128, BN = 128, BK = 32;
    __shared__ __attribute__((aligned(16))) bf16 As[2][BM][BK + 8];
    __shared__ __attribute__((aligned(16))) bf16 Bs[2][BN][BK + 8];

    const int tid  = threadIdx.x;
    const int lane = tid & 31;
    const int wave = tid >> 5;
    const int l16  = lane & 15;
    const int hi   = lane >> 4;           // 0 or 1 (half-wave select)
    const int wm   = (wave & 3) * 32;     // wave row offset inside tile
    const int wn   = (wave >> 2) * 64;    // wave col offset inside tile
    const int mBase = blockIdx.x * BM;
    const int nBase = blockIdx.y * BN;

    // per-thread DMA slots: 2 x 16B for A, 2 x 16B for B per K-step
    int ar[2], ac[2], bn_[2], bc[2];
#pragma unroll
    for (int s = 0; s < 2; s++) {
        int v = tid + s * 256;
        ar[s] = v >> 2;
        ac[s] = (v & 3) * 8;
        int n = nBase + ar[s];
        bn_[s] = (n < N) ? n : (N - 1);   // clamp: garbage cols never stored
        bc[s]  = ac[s];
    }

    v8f acc[2][4];
#pragma unroll
    for (int i = 0; i < 2; i++)
#pragma unroll
        for (int j = 0; j < 4; j++)
#pragma unroll
            for (int e = 0; e < 8; e++) acc[i][j][e] = 0.0f;

    // issue one K-step's tile DMA into buffer `buf`
    auto issue_tiles = [&](int buf, int k0) {
#pragma unroll
        for (int s = 0; s < 2; s++) {
            unsigned lds = LDS_OFF(&As[buf][ar[s]][ac[s]]);
            unsigned long long ga = (unsigned long long)(uintptr_t)
                &A[(size_t)(mBase + ar[s]) * K + k0 + ac[s]];
            asm volatile("global_load_async_to_lds_b128 %0, %1, off"
                         :: "v"(lds), "v"(ga) : "memory");
        }
#pragma unroll
        for (int s = 0; s < 2; s++) {
            unsigned lds = LDS_OFF(&Bs[buf][ar[s]][bc[s]]);
            unsigned long long ga = (unsigned long long)(uintptr_t)
                &B[(size_t)bn_[s] * K + k0 + bc[s]];
            asm volatile("global_load_async_to_lds_b128 %0, %1, off"
                         :: "v"(lds), "v"(ga) : "memory");
        }
    };

    const int nIter = K / BK;
    issue_tiles(0, 0);

    for (int it = 0; it < nIter; it++) {
        const int cur = it & 1;
        asm volatile("s_wait_asynccnt 0x0" ::: "memory"); // my DMA landed
        __syncthreads();                                  // everyone's DMA landed
        if (it + 1 < nIter) issue_tiles(cur ^ 1, (it + 1) * BK);

        // A fragments: lanes 0-15 K{0..7,16..23}, lanes 16-31 K{8..15,24..31}
        v16bf afr[2];
        const int ka = hi * 8;
#pragma unroll
        for (int i = 0; i < 2; i++)
            afr[i] = pack16(&As[cur][wm + i * 16 + l16][ka],
                            &As[cur][wm + i * 16 + l16][ka + 16]);
        // B fragments: lane n = l16, lanes 0-15 K{0..7,16..23}, lanes 16-31 K{8..15,24..31}
        v16bf bfr[4];
        const int kb = hi * 8;
#pragma unroll
        for (int j = 0; j < 4; j++)
            bfr[j] = pack16(&Bs[cur][wn + j * 16 + l16][kb],
                            &Bs[cur][wn + j * 16 + l16][kb + 16]);

#pragma unroll
        for (int i = 0; i < 2; i++)
#pragma unroll
            for (int j = 0; j < 4; j++)
                acc[i][j] = __builtin_amdgcn_wmma_f32_16x16x32_bf16(
                    false, afr[i], false, bfr[j], (short)0, acc[i][j],
                    false, false);
        __syncthreads();  // done reading buf[cur] before it is refilled
    }

    // C layout: VGPR g -> row (hi*8 + g), col = l16
#pragma unroll
    for (int i = 0; i < 2; i++)
#pragma unroll
        for (int j = 0; j < 4; j++)
#pragma unroll
            for (int g = 0; g < 8; g++) {
                int row = mBase + wm + i * 16 + hi * 8 + g;
                int col = nBase + wn + j * 16 + l16;
                if (col < N) C[(size_t)row * N + col] = (OutT)acc[i][j][g];
            }
}

// ---------------------------------------------------------------------------
// RMSNorm over C columns of a (possibly wider-stride) bf16 row; fp32 math.
// One block (256 threads = 8 waves) per row.
// ---------------------------------------------------------------------------
__global__ __launch_bounds__(256)
void rmsnorm_bf16(const bf16* __restrict__ x, int xStride,
                  const float* __restrict__ w,
                  bf16* __restrict__ y, int yStride, int C) {
    const int row = blockIdx.x;
    const bf16* xr = x + (size_t)row * xStride;
    bf16* yr = y + (size_t)row * yStride;

    float ss = 0.0f;
    for (int c = threadIdx.x; c < C; c += blockDim.x) {
        float v = (float)xr[c];
        ss += v * v;
    }
#pragma unroll
    for (int m = 16; m >= 1; m >>= 1) ss += __shfl_xor(ss, m, 32);
    __shared__ float red[8];
    if ((threadIdx.x & 31) == 0) red[threadIdx.x >> 5] = ss;
    __syncthreads();
    float tot = 0.0f;
#pragma unroll
    for (int i = 0; i < 8; i++) tot += red[i];
    const float sc = rsqrtf(tot / (float)C + 1e-6f);
    for (int c = threadIdx.x; c < C; c += blockDim.x)
        yr[c] = (bf16)((float)xr[c] * sc * w[c]);
}

// ---------------------------------------------------------------------------
// RoPE helper
// ---------------------------------------------------------------------------
__device__ __forceinline__ float rope_val(float x, float other, int i, int s) {
    const int j = i & 31;                               // frequency index
    const float inv = __expf(-((float)(2 * j) / 64.0f) * 9.210340372f); // ln(1e4)
    const float ang = (float)s * inv;
    const float c = __cosf(ang), sn = __sinf(ang);
    const float rot = (i < 32) ? -other : other;
    return x * c + rot * sn;
}

// q [B*S][NH*192] -> qf [B*NH][S][192], RoPE on last 64, pre-scaled by SCALE
__global__ void build_qf(const bf16* __restrict__ q, bf16* __restrict__ qf,
                         float scale) {
    const int row = blockIdx.x;            // b*S + s
    const int h   = blockIdx.y;
    const int d   = threadIdx.x;           // 0..191
    if (d >= DQK) return;
    const int b = row / SEQ, s = row % SEQ;
    const size_t inBase = (size_t)row * (NHEADS * DQK) + (size_t)h * DQK;
    float val = (float)q[inBase + d];
    if (d >= DNOPE) {
        const int i = d - DNOPE;
        const float other =
            (float)q[inBase + DNOPE + ((i < 32) ? i + 32 : i - 32)];
        val = rope_val(val, other, i, s);
    }
    qf[(((size_t)b * NHEADS + h) * SEQ + s) * DQK + d] = (bf16)(val * scale);
}

// kvdec [B*S][NH*256] + kvraw [B*S][576] -> kf [B*NH][S][192], v [B*NH][S][128]
__global__ void build_kfv(const bf16* __restrict__ kvdec,
                          const bf16* __restrict__ kvraw,
                          bf16* __restrict__ kf, bf16* __restrict__ v) {
    const int row = blockIdx.x;
    const int h   = blockIdx.y;
    const int d   = threadIdx.x;           // 0..255
    const int b = row / SEQ, s = row % SEQ;
    const size_t headBase = ((size_t)b * NHEADS + h) * SEQ + s;
    const size_t decBase  = (size_t)row * KV_OUT + (size_t)h * (DNOPE + DVAL);
    if (d < DNOPE) {
        kf[headBase * DQK + d]  = kvdec[decBase + d];
        v[headBase * DVAL + d]  = kvdec[decBase + DNOPE + d];
    } else if (d < DQK) {
        const int i = d - DNOPE;
        const float x = (float)kvraw[(size_t)row * KV_IN + KVLORA + i];
        const float other =
            (float)kvraw[(size_t)row * KV_IN + KVLORA + ((i < 32) ? i + 32 : i - 32)];
        kf[headBase * DQK + d] = (bf16)rope_val(x, other, i, s);
    }
}

// ---------------------------------------------------------------------------
// Flash attention (causal). Block = (64-query tile, b*NH). 128 threads = 4
// waves, each wave owns 16 query rows. Scores via WMMA (K=192), online
// softmax with shfl_xor row reductions, P staged through per-wave LDS,
// V transposed in LDS so PV B-fragments are contiguous.
// ---------------------------------------------------------------------------
__global__ __launch_bounds__(128)
void flash_attn(const bf16* __restrict__ qf, const bf16* __restrict__ kf,
                const bf16* __restrict__ v, bf16* __restrict__ out) {
    constexpr int BQ = 64, BKV = 64;
    __shared__ __attribute__((aligned(16))) bf16 Qs[BQ][DQK + 8];
    __shared__ __attribute__((aligned(16))) bf16 Ks[BKV][DQK + 8];
    __shared__ __attribute__((aligned(16))) bf16 Vs[DVAL][BKV + 8];   // [d][key]
    __shared__ __attribute__((aligned(16))) bf16 Ps[4][16][BKV + 8];  // per-wave P

    const int qi  = blockIdx.x;
    const int bh  = blockIdx.y;
    const int b   = bh / NHEADS, h = bh % NHEADS;
    const int tid = threadIdx.x, lane = tid & 31, wave = tid >> 5;
    const int l16 = lane & 15, hs = lane >> 4;
    const int q0  = qi * BQ;
    const int m0  = wave * 16;

    // load Q tile
    const size_t baseQ = ((size_t)bh * SEQ + q0) * DQK;
#pragma unroll
    for (int vv = tid; vv < BQ * DQK / 8; vv += 128) {
        int r = vv / (DQK / 8), c = (vv % (DQK / 8)) * 8;
        *(bf16x8*)&Qs[r][c] = *(const bf16x8*)&qf[baseQ + (size_t)r * DQK + c];
    }

    v8f accO[8];
#pragma unroll
    for (int t = 0; t < 8; t++)
#pragma unroll
        for (int e = 0; e < 8; e++) accO[t][e] = 0.0f;
    float mSt[8], lSt[8];
#pragma unroll
    for (int g = 0; g < 8; g++) { mSt[g] = -3.0e38f; lSt[g] = 0.0f; }

    for (int kj = 0; kj <= qi; kj++) {
        __syncthreads();
        const int k0 = kj * BKV;
        const size_t baseK = ((size_t)bh * SEQ + k0) * DQK;
#pragma unroll
        for (int vv = tid; vv < BKV * DQK / 8; vv += 128) {
            int r = vv / (DQK / 8), c = (vv % (DQK / 8)) * 8;
            *(bf16x8*)&Ks[r][c] = *(const bf16x8*)&kf[baseK + (size_t)r * DQK + c];
        }
        const size_t baseV = ((size_t)bh * SEQ + k0) * DVAL;
#pragma unroll
        for (int vv = tid; vv < BKV * DVAL / 8; vv += 128) {
            int r = vv / (DVAL / 8), c = (vv % (DVAL / 8)) * 8;
            bf16x8 vd = *(const bf16x8*)&v[baseV + (size_t)r * DVAL + c];
#pragma unroll
            for (int e = 0; e < 8; e++) Vs[c + e][r] = vd[e];  // transpose
        }
        __syncthreads();

        // ---- scores: S = Q * K^T (16x64 strip per wave) ----
        v8f accS[4];
#pragma unroll
        for (int t = 0; t < 4; t++)
#pragma unroll
            for (int e = 0; e < 8; e++) accS[t][e] = 0.0f;

#pragma unroll
        for (int kc = 0; kc < DQK; kc += 32) {
            const int ka = kc + hs * 8;
            v16bf afr = pack16(&Qs[m0 + l16][ka], &Qs[m0 + l16][ka + 16]);
            const int kb = kc + hs * 8;
#pragma unroll
            for (int t = 0; t < 4; t++) {
                v16bf bfr = pack16(&Ks[t * 16 + l16][kb],
                                   &Ks[t * 16 + l16][kb + 16]);
                accS[t] = __builtin_amdgcn_wmma_f32_16x16x32_bf16(
                    false, afr, false, bfr, (short)0, accS[t], false, false);
            }
        }

        // causal mask on the diagonal block
        if (kj == qi) {
#pragma unroll
            for (int t = 0; t < 4; t++)
#pragma unroll
                for (int g = 0; g < 8; g++) {
                    int qIdx = m0 + hs * 8 + g;
                    int kIdx = t * 16 + l16;
                    if (kIdx > qIdx) accS[t][g] = -3.0e38f;
                }
        }

        // ---- online softmax (row = hs*8 + g within wave strip) ----
#pragma unroll
        for (int g = 0; g < 8; g++) {
            float r = accS[0][g];
#pragma unroll
            for (int t = 1; t < 4; t++) r = fmaxf(r, accS[t][g]);
#pragma unroll
            for (int mm = 8; mm >= 1; mm >>= 1)
                r = fmaxf(r, __shfl_xor(r, mm, 32));
            const float mNew = fmaxf(mSt[g], r);
            const float alpha = __expf(mSt[g] - mNew);
            float rs = 0.0f;
#pragma unroll
            for (int t = 0; t < 4; t++) {
                float p = __expf(accS[t][g] - mNew);
                accS[t][g] = p;
                rs += p;
            }
#pragma unroll
            for (int mm = 8; mm >= 1; mm >>= 1) rs += __shfl_xor(rs, mm, 32);
            lSt[g] = lSt[g] * alpha + rs;
            mSt[g] = mNew;
#pragma unroll
            for (int t = 0; t < 8; t++) accO[t][g] *= alpha;
        }

        // ---- stage P through per-wave LDS (C layout -> A layout) ----
#pragma unroll
        for (int t = 0; t < 4; t++)
#pragma unroll
            for (int g = 0; g < 8; g++)
                Ps[wave][hs * 8 + g][t * 16 + l16] = (bf16)accS[t][g];
        asm volatile("s_wait_dscnt 0" ::: "memory");  // same-wave DS RAW

        // ---- O += P * V ----
#pragma unroll
        for (int kc = 0; kc < 2; kc++) {
            const int ka = kc * 32 + hs * 8;
            v16bf pfr = pack16(&Ps[wave][l16][ka], &Ps[wave][l16][ka + 16]);
            const int kb = kc * 32 + hs * 8;
#pragma unroll
            for (int dt = 0; dt < 8; dt++) {
                v16bf bfr = pack16(&Vs[dt * 16 + l16][kb],
                                   &Vs[dt * 16 + l16][kb + 16]);
                accO[dt] = __builtin_amdgcn_wmma_f32_16x16x32_bf16(
                    false, pfr, false, bfr, (short)0, accO[dt], false, false);
            }
        }
    }

    // epilogue: O /= l, write attn [B*S][NH*128]
    const size_t outRow0 = (size_t)b * SEQ + q0;
#pragma unroll
    for (int dt = 0; dt < 8; dt++)
#pragma unroll
        for (int g = 0; g < 8; g++) {
            int r = m0 + hs * 8 + g;
            float o = accO[dt][g] / lSt[g];
            out[(outRow0 + r) * (NHEADS * DVAL) + h * DVAL + dt * 16 + l16] =
                (bf16)o;
        }
}

// ---------------------------------------------------------------------------
// Launcher
// ---------------------------------------------------------------------------
extern "C" void kernel_launch(void* const* d_in, const int* in_sizes, int n_in,
                              void* d_out, int out_size, void* d_ws,
                              size_t ws_size, hipStream_t stream) {
    const float* hidden    = (const float*)d_in[0];  // [2,2048,2048]
    const float* wq_a      = (const float*)d_in[1];  // [1536,2048]
    const float* q_norm_w  = (const float*)d_in[2];  // [1536]
    const float* wq_b      = (const float*)d_in[3];  // [3072,1536]
    const float* wkv_a     = (const float*)d_in[4];  // [576,2048]
    const float* kv_norm_w = (const float*)d_in[5];  // [512]
    const float* wkv_b     = (const float*)d_in[6];  // [4096,512]
    const float* wo        = (const float*)d_in[7];  // [2048,2048]
    float* out = (float*)d_out;                      // [4096,2048]

    char* ws = (char*)d_ws;
    size_t off = 0;
    auto alloc = [&](size_t bytes) -> char* {
        char* p = ws + off;
        off += (bytes + 255) & ~(size_t)255;
        return p;
    };

    bf16* hbf   = (bf16*)alloc((size_t)MROWS * H_DIM * 2);   // later: attn
    bf16* wbuf  = (bf16*)alloc((size_t)3072 * QLORA * 2);    // largest weight
    bf16* qlora = (bf16*)alloc((size_t)MROWS * QLORA * 2);   // later: qf (lo)
    bf16* qn    = (bf16*)alloc((size_t)MROWS * QLORA * 2);   // later: qf (hi)
    bf16* qbf   = (bf16*)alloc((size_t)MROWS * NHEADS * DQK * 2); // later: kf
    bf16* kvbf  = (bf16*)alloc((size_t)MROWS * KV_IN * 2);
    bf16* kvn   = (bf16*)alloc((size_t)MROWS * KVLORA * 2);
    bf16* kvdec = (bf16*)alloc((size_t)MROWS * KV_OUT * 2);
    bf16* vbuf  = (bf16*)alloc((size_t)MROWS * NHEADS * DVAL * 2);
    bf16* qfb  = qlora;  // 2*12.58MB region, dead after q-up GEMM
    bf16* kfb  = qbf;    // dead after build_qf
    bf16* attn = hbf;    // dead after kv-down GEMM

    const float SCALE = 1.0f / sqrtf((float)DQK);
    auto castN = [&](const float* src, bf16* dst, size_t n) {
        int blocks = (int)((n + 2047) / 2048);
        cast_f32_to_bf16<<<blocks, 256, 0, stream>>>(src, dst, (int)n);
    };

    // 1) hidden -> bf16
    castN(hidden, hbf, (size_t)MROWS * H_DIM);

    // 2) q_lora = hidden @ wq_a^T
    castN(wq_a, wbuf, (size_t)QLORA * H_DIM);
    gemm_bf16_wmma<bf16><<<dim3(MROWS / 128, QLORA / 128), 256, 0, stream>>>(
        hbf, wbuf, qlora, MROWS, QLORA, H_DIM);

    // 3) rmsnorm(q_lora)
    rmsnorm_bf16<<<MROWS, 256, 0, stream>>>(qlora, QLORA, q_norm_w, qn, QLORA,
                                            QLORA);

    // 4) q = qn @ wq_b^T
    castN(wq_b, wbuf, (size_t)NHEADS * DQK * QLORA);
    gemm_bf16_wmma<bf16><<<dim3(MROWS / 128, (NHEADS * DQK) / 128), 256, 0,
                           stream>>>(qn, wbuf, qbf, MROWS, NHEADS * DQK, QLORA);

    // 5) kv = hidden @ wkv_a^T   (N = 576, ragged tile handled in kernel)
    castN(wkv_a, wbuf, (size_t)KV_IN * H_DIM);
    gemm_bf16_wmma<bf16><<<dim3(MROWS / 128, (KV_IN + 127) / 128), 256, 0,
                           stream>>>(hbf, wbuf, kvbf, MROWS, KV_IN, H_DIM);

    // 6) rmsnorm(kv_c) -- first 512 cols of the 576-wide rows
    rmsnorm_bf16<<<MROWS, 256, 0, stream>>>(kvbf, KV_IN, kv_norm_w, kvn, KVLORA,
                                            KVLORA);

    // 7) kv_dec = kvn @ wkv_b^T
    castN(wkv_b, wbuf, (size_t)KV_OUT * KVLORA);
    gemm_bf16_wmma<bf16><<<dim3(MROWS / 128, KV_OUT / 128), 256, 0, stream>>>(
        kvn, wbuf, kvdec, MROWS, KV_OUT, KVLORA);

    // 8) assemble qf (RoPE + fold SCALE), kf, v in [B*NH][S][D] layout
    build_qf<<<dim3(MROWS, NHEADS), 192, 0, stream>>>(qbf, qfb, SCALE);
    build_kfv<<<dim3(MROWS, NHEADS), 256, 0, stream>>>(kvdec, kvbf, kfb, vbuf);

    // 9) causal flash attention
    flash_attn<<<dim3(SEQ / 64, BATCH * NHEADS), 128, 0, stream>>>(qfb, kfb,
                                                                   vbuf, attn);

    // 10) out = attn @ wo^T   (fp32 output)
    castN(wo, wbuf, (size_t)H_DIM * H_DIM);
    gemm_bf16_wmma<float><<<dim3(MROWS / 128, H_DIM / 128), 256, 0, stream>>>(
        attn, wbuf, out, MROWS, H_DIM, H_DIM);
}
